// VolumePreservingFlows_44659069944367
// MI455X (gfx1250) — matrix operand order repaired and checked by reading, not compiled, not run on hardware
//
#include <hip/hip_runtime.h>
#include <hip/hip_bf16.h>
#include <cstdint>

// ---------------- types ----------------
typedef __attribute__((ext_vector_type(16))) _Float16 v16h;
typedef __attribute__((ext_vector_type(8)))  _Float16 v8h;
typedef __attribute__((ext_vector_type(8)))  float    v8f;
typedef __attribute__((ext_vector_type(4)))  float    v4f;

// ---------------- problem constants ----------------
#define BATCH  65536
#define LDIM   256
#define HIN    300
#define K0PAD  320      // HIN padded up to multiple of 32 (zeros)
#define NFLOWS 4

// ---------------- tiling ----------------
#define WAVES       4
#define ROWS_PER_WG (WAVES * 16)
#define ZSTR 260        // floats per z row in LDS  (bank advance 4/row)
#define VSTR 264        // halfs  per v row in LDS  (bank advance 4/row)
#define LDS_BYTES (WAVES * 16 * ZSTR * 4 + WAVES * 16 * VSTR * 2)  // 100352 B

// ---- CDNA5 async LDS<->global helpers (ASYNCcnt-tracked, no VGPR staging) ----
__device__ __forceinline__ void async_g2lds_b128(uint32_t lds_addr, uint64_t gaddr) {
  asm volatile("global_load_async_to_lds_b128 %0, %1, off"
               :: "v"(lds_addr), "v"(gaddr) : "memory");
}
__device__ __forceinline__ void async_lds2g_b128(uint64_t gaddr, uint32_t lds_addr) {
  asm volatile("global_store_async_from_lds_b128 %0, %1, off"
               :: "v"(gaddr), "v"(lds_addr) : "memory");
}
__device__ __forceinline__ void wait_asynccnt0() {
  asm volatile("s_wait_asynccnt 0x0" ::: "memory");
}
__device__ __forceinline__ void wait_dscnt0() {
  asm volatile("s_wait_dscnt 0x0" ::: "memory");
}

// ============================================================
// Prep: convert weights to f16 (W0 zero-padded to K=320)
// ============================================================
__global__ void vpflow_prep_weights(const float* __restrict__ W0,
                                    const float* __restrict__ Ws,
                                    _Float16* __restrict__ W0h,
                                    _Float16* __restrict__ Wsh) {
  int i = blockIdx.x * blockDim.x + threadIdx.x;
  if (i < LDIM * K0PAD) {
    int n = i / K0PAD, k = i % K0PAD;
    W0h[i] = (k < HIN) ? (_Float16)W0[n * HIN + k] : (_Float16)0.0f;
  }
  if (i < (NFLOWS - 1) * LDIM * LDIM) {
    Wsh[i] = (_Float16)Ws[i];
  }
}

// ============================================================
// Fused kernel: all 4 flows + Householder updates, z LDS-resident
// ============================================================
union AFrag { v16h v; v8h h[2]; _Float16 e[16]; };

__global__ __launch_bounds__(WAVES * 32) void vpflow_kernel(
    const float* __restrict__ z_in, const float* __restrict__ h,
    const float* __restrict__ b0,   const float* __restrict__ bs,
    const _Float16* __restrict__ W0h, const _Float16* __restrict__ Wsh,
    float* __restrict__ z_out) {
  extern __shared__ char smem[];
  const int tid  = threadIdx.x;
  const int wave = tid >> 5;
  const int lane = tid & 31;
  const int r    = lane & 15;   // row (A,B frag) / col (D frag) within tile
  const int hi   = lane >> 4;   // K-half selector per ISA WMMA layouts

  float*    zb = (float*)smem + wave * (16 * ZSTR);
  _Float16* vb = (_Float16*)((float*)smem + WAVES * 16 * ZSTR) + wave * (16 * VSTR);

  const int rowBase = blockIdx.x * ROWS_PER_WG + wave * 16;

  // ---- fire async z staging: lands in LDS while flow-0 GEMM runs ----
  {
    const float* zg = z_in + (size_t)rowBase * LDIM;
#pragma unroll
    for (int i = lane; i < 16 * (LDIM / 4); i += 32) {
      int row = i >> 6, c4 = i & 63;
      async_g2lds_b128((uint32_t)(uintptr_t)(zb + row * ZSTR + c4 * 4),
                       (uint64_t)(uintptr_t)(zg + row * LDIM + c4 * 4));
    }
  }

  // ---- flow 0 GEMM: v = h @ W0^T + b0  (K = 300 padded to 320) ----
  {
    AFrag a[K0PAD / 32];
    const float* hrow = h + (size_t)(rowBase + r) * HIN;
#pragma unroll
    for (int c = 0; c < K0PAD / 32; ++c) {
      const int kA = c * 32 + hi * 8;        // A layout: lo-half K group
      const int kB = c * 32 + 16 + hi * 8;   // A layout: hi-half K group
#pragma unroll
      for (int j = 0; j < 8; ++j) {
        int ka = kA + j, kb = kB + j;
        a[c].e[j]     = (ka < HIN) ? (_Float16)__builtin_nontemporal_load(hrow + ka)
                                   : (_Float16)0.0f;
        a[c].e[8 + j] = (kb < HIN) ? (_Float16)__builtin_nontemporal_load(hrow + kb)
                                   : (_Float16)0.0f;
      }
    }
#pragma unroll
    for (int nt = 0; nt < 16; ++nt) {
      v8f acc = {};
      const _Float16* wrow = W0h + (size_t)(nt * 16 + r) * K0PAD + hi * 8;
#pragma unroll
      for (int c = 0; c < K0PAD / 32; ++c) {
        v16h bf = __builtin_shufflevector(*(const v8h*)(wrow + c * 32), *(const v8h*)(wrow + c * 32 + 16), 0, 1, 2, 3, 4, 5, 6, 7, 8, 9, 10, 11, 12, 13, 14, 15);   // B: lane n holds K groups 8h..8h+7 and 16+8h..16+8h+7
        acc = __builtin_amdgcn_wmma_f32_16x16x32_f16(
            false, a[c].v, false, bf, (short)0, acc, false, false);
      }
      float bias = b0[nt * 16 + r];                // D layout: lane holds col n=r
#pragma unroll
      for (int j = 0; j < 8; ++j)
        vb[(j + hi * 8) * VSTR + nt * 16 + r] = (_Float16)(acc[j] + bias);
    }
  }
  __syncthreads();

  // ---- flows: Householder(z, v_f) then v_{f+1} = v_f @ W^T + b ----
  AFrag vf[LDIM / 32];
#pragma unroll
  for (int f = 0; f < NFLOWS; ++f) {
    // reload v as A-fragments from LDS (serves both Householder & next GEMM)
    const _Float16* vrow = vb + r * VSTR;
#pragma unroll
    for (int c = 0; c < LDIM / 32; ++c) {
      vf[c].h[0] = *(const v8h*)(vrow + c * 32 + hi * 8);
      vf[c].h[1] = *(const v8h*)(vrow + c * 32 + 16 + hi * 8);
    }

    // first use of z: make sure the async staging has landed
    if (f == 0) wait_asynccnt0();

    // Householder: vz = v.z ; vv = v.v ; z -= (2*vz/vv) * v
    float vz = 0.f, vv = 0.f;
    float* zrow = zb + r * ZSTR;
#pragma unroll
    for (int c = 0; c < LDIM / 32; ++c)
#pragma unroll
      for (int g = 0; g < 2; ++g) {
        int kb = c * 32 + g * 16 + hi * 8;
        v4f z0 = *(v4f*)(zrow + kb);
        v4f z1 = *(v4f*)(zrow + kb + 4);
#pragma unroll
        for (int j = 0; j < 4; ++j) {
          float va = (float)vf[c].e[g * 8 + j];
          float wb = (float)vf[c].e[g * 8 + 4 + j];
          vz += va * z0[j] + wb * z1[j];
          vv += va * va + wb * wb;
        }
      }
    // lanes l and l^16 hold disjoint halves of row r: one xor-16 reduce
    vz += __shfl_xor(vz, 16, 32);
    vv += __shfl_xor(vv, 16, 32);
    const float coef = 2.0f * vz / vv;
#pragma unroll
    for (int c = 0; c < LDIM / 32; ++c)
#pragma unroll
      for (int g = 0; g < 2; ++g) {
        int kb = c * 32 + g * 16 + hi * 8;
        v4f z0 = *(v4f*)(zrow + kb);
        v4f z1 = *(v4f*)(zrow + kb + 4);
#pragma unroll
        for (int j = 0; j < 4; ++j) {
          z0[j] -= coef * (float)vf[c].e[g * 8 + j];
          z1[j] -= coef * (float)vf[c].e[g * 8 + 4 + j];
        }
        *(v4f*)(zrow + kb)     = z0;
        *(v4f*)(zrow + kb + 4) = z1;
      }

    if (f < NFLOWS - 1) {
      __syncthreads();
      const _Float16* W   = Wsh + (size_t)f * (LDIM * LDIM);
      const float* bias_p = bs + f * LDIM;
#pragma unroll
      for (int nt = 0; nt < 16; ++nt) {
        v8f acc = {};
        const _Float16* wrow = W + (size_t)(nt * 16 + r) * LDIM + hi * 8;
#pragma unroll
        for (int c = 0; c < LDIM / 32; ++c) {
          v16h bf = __builtin_shufflevector(*(const v8h*)(wrow + c * 32), *(const v8h*)(wrow + c * 32 + 16), 0, 1, 2, 3, 4, 5, 6, 7, 8, 9, 10, 11, 12, 13, 14, 15);
          acc = __builtin_amdgcn_wmma_f32_16x16x32_f16(
              false, vf[c].v, false, bf, (short)0, acc, false, false);
        }
        float bias = bias_p[nt * 16 + r];
#pragma unroll
        for (int j = 0; j < 8; ++j)
          vb[(j + hi * 8) * VSTR + nt * 16 + r] = (_Float16)(acc[j] + bias);
      }
      __syncthreads();
    }
  }

  // ---- write z back: async LDS -> global ----
  __syncthreads();          // emits s_wait_dscnt: LDS z updates visible
  wait_dscnt0();            // explicit, for the async engine's LDS reads
  {
    float* zg = z_out + (size_t)rowBase * LDIM;
#pragma unroll
    for (int i = lane; i < 16 * (LDIM / 4); i += 32) {
      int row = i >> 6, c4 = i & 63;
      async_lds2g_b128((uint64_t)(uintptr_t)(zg + row * LDIM + c4 * 4),
                       (uint32_t)(uintptr_t)(zb + row * ZSTR + c4 * 4));
    }
    wait_asynccnt0();       // S_ENDPGM would also wait-idle; explicit is clearer
  }
}

// ============================================================
// Launch
// ============================================================
extern "C" void kernel_launch(void* const* d_in, const int* in_sizes, int n_in,
                              void* d_out, int out_size, void* d_ws, size_t ws_size,
                              hipStream_t stream) {
  const float* z  = (const float*)d_in[0];
  const float* h  = (const float*)d_in[1];
  const float* W0 = (const float*)d_in[2];
  const float* b0 = (const float*)d_in[3];
  const float* Ws = (const float*)d_in[4];
  const float* bs = (const float*)d_in[5];
  float* z_out    = (float*)d_out;

  _Float16* W0h = (_Float16*)d_ws;                 // 256*320 halfs
  _Float16* Wsh = W0h + LDIM * K0PAD;              // 3*256*256 halfs

  const int prepN = (NFLOWS - 1) * LDIM * LDIM;    // covers both arrays
  vpflow_prep_weights<<<(prepN + 255) / 256, 256, 0, stream>>>(W0, Ws, W0h, Wsh);

  hipFuncSetAttribute((const void*)vpflow_kernel,
                      hipFuncAttributeMaxDynamicSharedMemorySize, LDS_BYTES);

  vpflow_kernel<<<BATCH / ROWS_PER_WG, WAVES * 32, LDS_BYTES, stream>>>(
      z, h, b0, bs, W0h, Wsh, z_out);
}
